// GCNEncoder_28948079575789
// MI455X (gfx1250) — matrix operand order repaired and checked by reading, not compiled, not run on hardware
//
#include <hip/hip_runtime.h>

typedef __attribute__((ext_vector_type(16))) __bf16 v16bf;
typedef __attribute__((ext_vector_type(8)))  float  v8f;

// =============== CSR-by-destination build ===============
// cnt[i] starts at 1 (self loop), then +1 per in-edge. Also zeroes the global cursor.
__global__ __launch_bounds__(256) void cnt_init(int* cnt, int* gcur, int nN) {
    int i = blockIdx.x * 256 + threadIdx.x;
    if (i < nN) cnt[i] = 1;
    if (blockIdx.x == 0 && threadIdx.x == 0) *gcur = 0;
}

__global__ __launch_bounds__(256) void cnt_accum(const int* __restrict__ dst, int* cnt, int nE) {
    int e = blockIdx.x * 256 + threadIdx.x;
    if (e < nE) atomicAdd(&cnt[dst[e]], 1);
}

// dinv = rsqrt(deg); allocate each node's disjoint CSR segment off one global cursor
// (no prefix scan needed: segment order is arbitrary, gather only needs disjointness).
__global__ __launch_bounds__(256) void node_setup(const int* __restrict__ cnt, float* dinv,
                                                  int* rowstart, int* cursor, int* gcur, int nN) {
    int i = blockIdx.x * 256 + threadIdx.x;
    if (i < nN) {
        int c = cnt[i];
        dinv[i] = rsqrtf((float)c);
        int rs = atomicAdd(gcur, c);
        rowstart[i] = rs;
        cursor[i]   = rs;
    }
}

// One thread per edge (incl. self loops): write packed {src, norm} record into dst's segment.
__global__ __launch_bounds__(256) void scatter_edges(const int* __restrict__ src,
                                                     const int* __restrict__ dst,
                                                     const float* __restrict__ dinv,
                                                     int* cursor, int2* __restrict__ csr,
                                                     int nE, int nN) {
    int e = blockIdx.x * 256 + threadIdx.x;
    if (e >= nE + nN) return;
    int s, d;
    if (e < nE) { s = src[e]; d = dst[e]; }
    else        { s = d = e - nE; }
    float nrm = dinv[s] * dinv[d];
    int pos = atomicAdd(&cursor[d], 1);
    csr[pos] = make_int2(s, __float_as_int(nrm));
}

// =============== WMMA tiled GEMM: Hout[N, NC] = X[N, K] @ W[K, NC] ===============
// bf16 staged in LDS, f32 accumulate via v_wmma_f32_16x16x32_bf16.
// 256 threads = 8 waves; each wave owns a 16-row stripe of a 128-row tile.
template<int K, int NC>
__global__ __launch_bounds__(256) void gemm_bf16(const float* __restrict__ X,
                                                 const float* __restrict__ W,
                                                 float* __restrict__ Hout, int Nrows) {
    constexpr int KP  = K + 8;       // pad to break LDS bank conflicts (keeps pairs aligned)
    constexpr int NCT = NC / 16;     // column tiles per wave
    extern __shared__ __bf16 smem[];
    __bf16* lda = smem;              // [128][KP]  row-major X tile
    __bf16* ldb = smem + 128 * KP;   // [NC][KP]   column-major W tile

    const int tid   = threadIdx.x;
    const int r0    = blockIdx.x * 128;
    const bool full = (r0 + 128) <= Nrows;

    for (int t = tid; t < 128 * K; t += 256) {          // stage X (fp32 -> bf16)
        int row = t / K, k = t - row * K;
        int gr = r0 + row;
        float v = (full || gr < Nrows) ? X[(size_t)gr * K + k] : 0.0f;
        lda[row * KP + k] = (__bf16)v;
    }
    for (int t = tid; t < K * NC; t += 256) {           // stage W column-major
        int k = t / NC, n = t - k * NC;
        ldb[n * KP + k] = (__bf16)W[(size_t)k * NC + n];
    }
    __syncthreads();

    const int wave = tid >> 5;
    const int lane = tid & 31;
    const int hlf  = lane >> 4;
    const int m16  = lane & 15;
    const int wrow = wave * 16;

    v8f acc[NCT];
    #pragma unroll
    for (int i = 0; i < NCT; i++) acc[i] = {};

    const __bf16* arow = lda + (wrow + m16) * KP;

    for (int ks = 0; ks < K / 32; ks++) {
        const int kb = ks * 32;
        v16bf a;                                        // 16-bit A 16x32 ISA layout
        #pragma unroll
        for (int i = 0; i < 8; i++) {
            int kk = kb + ((i >> 2) << 4) + (hlf << 3) + ((i & 3) << 1);
            a[2 * i]     = arow[kk];
            a[2 * i + 1] = arow[kk + 1];
        }
        #pragma unroll
        for (int ct = 0; ct < NCT; ct++) {              // B: same K order as A (16-bit ISA layout)
            const __bf16* bcol = ldb + (ct * 16 + m16) * KP + kb + (hlf << 3);
            v16bf b;
            #pragma unroll
            for (int i = 0; i < 8; i++) {
                int bk = ((i >> 2) << 4) + ((i & 3) << 1);
                b[2 * i]     = bcol[bk];
                b[2 * i + 1] = bcol[bk + 1];
            }
            acc[ct] = __builtin_amdgcn_wmma_f32_16x16x32_bf16(
                false, a, false, b, (short)0, acc[ct], false, false);
        }
    }

    const int rbase = r0 + wrow + hlf * 8;              // C/D: VGPR v -> M=v / v+8
    if (full) {
        #pragma unroll
        for (int ct = 0; ct < NCT; ct++) {
            float* cp = Hout + (size_t)rbase * NC + ct * 16 + m16;
            #pragma unroll
            for (int v = 0; v < 8; v++) cp[(size_t)v * NC] = acc[ct][v];
        }
    } else {
        #pragma unroll
        for (int ct = 0; ct < NCT; ct++) {
            float* cp = Hout + (size_t)rbase * NC + ct * 16 + m16;
            #pragma unroll
            for (int v = 0; v < 8; v++)
                if (rbase + v < Nrows) cp[(size_t)v * NC] = acc[ct][v];
        }
    }
}

// =============== gather aggregation (no atomics) ===============
// One wave per node (8 nodes / block); lane covers C/32 contiguous channels.
// Node id forced into an SGPR so rowstart/cnt and the packed {src,norm} records
// take the scalar path (s_load_b64 per edge); source rows are coalesced b128 (C=128) or b64 (C=64) loads.
// Bias init and optional ReLU are fused here.
template<int C, bool RELU>
__global__ __launch_bounds__(256) void agg_gather(const int* __restrict__ rowstart,
                                                  const int* __restrict__ cnt,
                                                  const int2* __restrict__ csr,
                                                  const float* __restrict__ Hin,
                                                  const float* __restrict__ bias,
                                                  float* __restrict__ Aout, int nN) {
    constexpr int VPL = C / 32;
    const int lane = threadIdx.x & 31;
    const int node = __builtin_amdgcn_readfirstlane((int)(blockIdx.x * 8) + (threadIdx.x >> 5));
    if (node >= nN) return;

    const int jb = rowstart[node];
    const int je = jb + cnt[node];

    float acc[VPL];
    #pragma unroll
    for (int i = 0; i < VPL; i++) acc[i] = bias[lane * VPL + i];

    for (int j = jb; j < je; ++j) {
        int2 r = csr[j];                       // scalar s_load_b64: {src, norm}
        int   s   = r.x;
        float nrm = __int_as_float(r.y);
        const float* hp = Hin + (size_t)s * C + lane * VPL;
        if constexpr (VPL == 4) {
            float4 h = *(const float4*)hp;
            acc[0] += h.x * nrm; acc[1] += h.y * nrm;
            acc[2] += h.z * nrm; acc[3] += h.w * nrm;
        } else {
            float2 h = *(const float2*)hp;
            acc[0] += h.x * nrm; acc[1] += h.y * nrm;
        }
    }

    float* ap = Aout + (size_t)node * C + lane * VPL;
    #pragma unroll
    for (int i = 0; i < VPL; i++) {
        float v = acc[i];
        if (RELU) v = fmaxf(v, 0.0f);
        ap[i] = v;
    }
}

// =============== launch ===============
extern "C" void kernel_launch(void* const* d_in, const int* in_sizes, int n_in,
                              void* d_out, int out_size, void* d_ws, size_t ws_size,
                              hipStream_t stream) {
    const float* x  = (const float*)d_in[0];   // [N, 256]
    const int*   ei = (const int*)d_in[1];     // [2, E]
    const float* W1 = (const float*)d_in[2];   // [256, 128]
    const float* b1 = (const float*)d_in[3];   // [128]
    const float* W2 = (const float*)d_in[4];   // [128, 64]
    const float* b2 = (const float*)d_in[5];   // [64]
    float* out = (float*)d_out;                // [N, 64]

    const int Nn = in_sizes[0] / 256;          // 50000
    const int nE = in_sizes[1] / 2;            // 800000
    const int nT = nE + Nn;                    // edges incl. self loops
    const int* src = ei;
    const int* dst = ei + nE;

    // workspace layout (4-byte words, 256-word aligned chunks)
    auto pad = [](size_t w) { return (w + 255) & ~(size_t)255; };
    char* wsb = (char*)d_ws;
    size_t o = 0;
    float* dinv     = (float*)(wsb + o * 4); o += pad(Nn);
    int*   cnt      = (int*)  (wsb + o * 4); o += pad(Nn);
    int*   rowstart = (int*)  (wsb + o * 4); o += pad(Nn);
    int*   cursor   = (int*)  (wsb + o * 4); o += pad(Nn);
    int*   gcur     = (int*)  (wsb + o * 4); o += 256;
    int2*  csr      = (int2*) (wsb + o * 4); o += pad((size_t)nT * 2);
    float* Hbuf     = (float*)(wsb + o * 4); o += (size_t)Nn * 128;
    float* AGG1     = (float*)(wsb + o * 4);

    const int nbN  = (Nn + 255) / 256;
    const int nbE  = (nE + 255) / 256;
    const int nbT  = (nT + 255) / 256;
    const int nbG  = (Nn + 127) / 128;
    const int nbAg = (Nn + 7) / 8;             // 8 node-waves per block

    // 1) CSR-by-dst build + normalization coefficients
    cnt_init     <<<nbN, 256, 0, stream>>>(cnt, gcur, Nn);
    cnt_accum    <<<nbE, 256, 0, stream>>>(dst, cnt, nE);
    node_setup   <<<nbN, 256, 0, stream>>>(cnt, dinv, rowstart, cursor, gcur, Nn);
    scatter_edges<<<nbT, 256, 0, stream>>>(src, dst, dinv, cursor, csr, nE, Nn);

    // 2) layer 1: H = X @ W1 (WMMA), AGG1 = relu(gather + b1)
    {
        constexpr int K = 256, NC = 128, KP = K + 8;
        size_t smem = (size_t)(128 * KP + NC * KP) * sizeof(__bf16);
        gemm_bf16<K, NC><<<nbG, 256, smem, stream>>>(x, W1, Hbuf, Nn);
    }
    agg_gather<128, true><<<nbAg, 256, 0, stream>>>(rowstart, cnt, csr, Hbuf, b1, AGG1, Nn);

    // 3) layer 2: H2 = AGG1 @ W2 (reuse Hbuf), out = gather + b2
    {
        constexpr int K = 128, NC = 64, KP = K + 8;
        size_t smem = (size_t)(128 * KP + NC * KP) * sizeof(__bf16);
        gemm_bf16<K, NC><<<nbG, 256, smem, stream>>>(AGG1, W2, Hbuf, Nn);
    }
    agg_gather<64, false><<<nbAg, 256, 0, stream>>>(rowstart, cnt, csr, Hbuf, b2, out, Nn);
}
